// Model_75548474736946
// MI455X (gfx1250) — compile-verified, matrix operand order checked by reading, not run on hardware
//
#include <hip/hip_runtime.h>
#include <hip/hip_bf16.h>

// ---------------------------------------------------------------------------
// Informer forward pass for MI455X (gfx1250, wave32, WMMA).
// All dense algebra runs on v_wmma_f32_16x16x32_f16; all LDS fragment traffic
// is vectorized (ds_load_b128 / ds_store_b128) by keeping tiles K-contiguous
// and using 16B-multiple, bank-conflict-free row strides (144B / 112B).
// ---------------------------------------------------------------------------

typedef __attribute__((ext_vector_type(16))) _Float16 v16h;
typedef __attribute__((ext_vector_type(8)))  float    v8f;

__device__ __forceinline__ v8f wmma16(v16h a, v16h b, v8f c) {
  return __builtin_amdgcn_wmma_f32_16x16x32_f16(false, a, false, b, (short)0, c,
                                                false, false);
}

// Load a 16x32-f16 A/B fragment whose per-lane K pattern is
// k = (h&7) + (h&8?16:0) + hi8  ->  two aligned 16B runs from one LDS row.
__device__ __forceinline__ v16h ld_frag(const _Float16* row, int hi8) {
  union { uint4 q[2]; v16h v; } u;
  u.q[0] = *(const uint4*)(row + hi8);
  u.q[1] = *(const uint4*)(row + 16 + hi8);
  return u.v;
}

// Convert 8 contiguous floats -> 8 halves, one b128 LDS store.
__device__ __forceinline__ void st8h(_Float16* dst, const float* src) {
  float4 a = *(const float4*)src;
  float4 b = *(const float4*)(src + 4);
  union { _Float16 h[8]; uint4 q; } u;
  u.h[0] = (_Float16)a.x; u.h[1] = (_Float16)a.y;
  u.h[2] = (_Float16)a.z; u.h[3] = (_Float16)a.w;
  u.h[4] = (_Float16)b.x; u.h[5] = (_Float16)b.y;
  u.h[6] = (_Float16)b.z; u.h[7] = (_Float16)b.w;
  *(uint4*)dst = u.q;
}

__device__ __forceinline__ void pack8(_Float16* dst, const float* s) {
  union { _Float16 h[8]; uint4 q; } u;
#pragma unroll
  for (int j = 0; j < 8; ++j) u.h[j] = (_Float16)s[j];
  *(uint4*)dst = u.q;
}

// Branch-free exact-GELU (erf via Abramowitz-Stegun 7.1.26, |err|<1.5e-7).
__device__ __forceinline__ float gelu_exact(float x) {
  float ax = fabsf(x);
  float t  = __builtin_amdgcn_rcpf(1.f + 0.3275911f * ax);
  float p  = t * (0.254829592f +
             t * (-0.284496736f +
             t * (1.421413741f +
             t * (-1.453152027f + t * 1.061405429f))));
  float er = 1.f - p * __expf(-x * x);
  er = (x < 0.f) ? -er : er;
  return 0.5f * x * (1.f + er);
}

// Build a 16x32 A-fragment directly from a global fp32 row (4x float4).
__device__ __forceinline__ v16h fragA_global(const float* p, int hi8) {
  float4 a = *(const float4*)(p + hi8);
  float4 b = *(const float4*)(p + hi8 + 4);
  float4 c = *(const float4*)(p + 16 + hi8);
  float4 d = *(const float4*)(p + 16 + hi8 + 4);
  v16h r;
  r[0]  = (_Float16)a.x; r[1]  = (_Float16)a.y; r[2]  = (_Float16)a.z; r[3]  = (_Float16)a.w;
  r[4]  = (_Float16)b.x; r[5]  = (_Float16)b.y; r[6]  = (_Float16)b.z; r[7]  = (_Float16)b.w;
  r[8]  = (_Float16)c.x; r[9]  = (_Float16)c.y; r[10] = (_Float16)c.z; r[11] = (_Float16)c.w;
  r[12] = (_Float16)d.x; r[13] = (_Float16)d.y; r[14] = (_Float16)d.z; r[15] = (_Float16)d.w;
  return r;
}

// ---------------------------------------------------------------------------
// GEMM: C[M,N] = act(A[M,K] @ W[K,N] + bias[N])
// 256 threads = 8 waves; block tile 128x64; wave tile 16x64 (4 accumulators
// share one A-fragment -> 4 WMMA per 2+8 LDS vector loads); k-step 32.
// Requires M%128==0, N%64==0, K%32==0 (true for every use here).
// ---------------------------------------------------------------------------
__global__ void gemm_kernel(const float* __restrict__ A,
                            const float* __restrict__ W,
                            const float* __restrict__ bias,
                            float* __restrict__ C,
                            int M, int N, int K, int act) {
  __shared__ _Float16 As[128][72];   // [m][k], 144B stride
  __shared__ _Float16 Ws[64][72];    // transposed: [n][k]
  const int tid  = threadIdx.x;
  const int wave = tid >> 5, lane = tid & 31;
  const int lo16 = lane & 15, hi8 = (lane & 16) ? 8 : 0;
  const int rowBase = blockIdx.x * 128;
  const int colBase = blockIdx.y * 64;
  const int arow = tid >> 1, acol = (tid & 1) * 16;
  const int wn = tid & 63, wkg = (tid >> 6) * 8;

  v8f acc[4] = {{}, {}, {}, {}};
  for (int k0 = 0; k0 < K; k0 += 32) {
    {  // stage A tile 128x32 (vector loads + b128 stores)
      const float* ap = A + (size_t)(rowBase + arow) * K + k0 + acol;
      st8h(&As[arow][acol], ap);
      st8h(&As[arow][acol + 8], ap + 8);
    }
    {  // stage W tile transposed: lane-coalesced global reads, b128 store
      float t[8];
#pragma unroll
      for (int j = 0; j < 8; ++j)
        t[j] = W[(size_t)(k0 + wkg + j) * N + colBase + wn];
      pack8(&Ws[wn][wkg], t);
    }
    __syncthreads();
    v16h af = ld_frag(&As[wave * 16 + lo16][0], hi8);
#pragma unroll
    for (int j = 0; j < 4; ++j) {
      v16h bf = ld_frag(&Ws[j * 16 + lo16][0], hi8);
      acc[j] = wmma16(af, bf, acc[j]);
    }
    __syncthreads();
  }
#pragma unroll
  for (int r = 0; r < 8; ++r) {
    int m  = rowBase + wave * 16 + r + hi8;
    int n0 = colBase + lo16;
    float* cp = C + (size_t)m * N + n0;
#pragma unroll
    for (int j = 0; j < 4; ++j) {
      float v = acc[j][r] + bias[n0 + 16 * j];
      if (act == 1) v = gelu_exact(v);
      cp[16 * j] = v;
    }
  }
}

// ---------------------------------------------------------------------------
// Circular conv1d (kernel=3, 512->512) as K=1536 WMMA GEMM with shifted
// A-tile gather. grid = (L/128, 512/64, B).
// ---------------------------------------------------------------------------
__global__ void conv_gemm_kernel(const float* __restrict__ X,
                                 const float* __restrict__ W,
                                 const float* __restrict__ bias,
                                 float* __restrict__ Y, int L) {
  __shared__ _Float16 As[128][72];
  __shared__ _Float16 Ws[64][72];
  const int N = 512, K = 1536, D = 512;
  const int tid  = threadIdx.x;
  const int wave = tid >> 5, lane = tid & 31;
  const int lo16 = lane & 15, hi8 = (lane & 16) ? 8 : 0;
  const int b = blockIdx.z;
  const int lBase   = blockIdx.x * 128;
  const int colBase = blockIdx.y * 64;
  const int arow = tid >> 1, acol = (tid & 1) * 16;
  const int wn = tid & 63, wkg = (tid >> 6) * 8;

  v8f acc[4] = {{}, {}, {}, {}};
  for (int k0 = 0; k0 < K; k0 += 32) {
    const int t = k0 >> 9;  // tap index, constant within a 32-wide k step
    {
      int l2 = lBase + arow + t - 1;
      if (l2 < 0) l2 += L;
      if (l2 >= L) l2 -= L;
      const float* ap = X + (size_t)(b * L + l2) * D + (k0 & 511) + acol;
      st8h(&As[arow][acol], ap);
      st8h(&As[arow][acol + 8], ap + 8);
    }
    {
      float tv[8];
#pragma unroll
      for (int j = 0; j < 8; ++j)
        tv[j] = W[(size_t)(k0 + wkg + j) * N + colBase + wn];
      pack8(&Ws[wn][wkg], tv);
    }
    __syncthreads();
    v16h af = ld_frag(&As[wave * 16 + lo16][0], hi8);
#pragma unroll
    for (int j = 0; j < 4; ++j) {
      v16h bf = ld_frag(&Ws[j * 16 + lo16][0], hi8);
      acc[j] = wmma16(af, bf, acc[j]);
    }
    __syncthreads();
  }
#pragma unroll
  for (int r = 0; r < 8; ++r) {
    int m  = lBase + wave * 16 + r + hi8;
    int n0 = colBase + lo16;
    float* yp = Y + (size_t)(b * L + m) * N + n0;
#pragma unroll
    for (int j = 0; j < 4; ++j) yp[16 * j] = acc[j][r] + bias[n0 + 16 * j];
  }
}

// ---------------------------------------------------------------------------
// Flash attention: WMMA scores + WMMA PV, online softmax, 8 waves x 16 rows.
// Q:[B,Lq,512] K,V:[B,Lk,512] (head h = columns h*64..h*64+63).
// grid = (Lq/128, H, B).
// ---------------------------------------------------------------------------
__global__ void attn_kernel(const float* __restrict__ Q,
                            const float* __restrict__ Kb,
                            const float* __restrict__ Vb,
                            float* __restrict__ O,
                            int Lq, int Lk, float scale, int causal) {
  __shared__ _Float16 Kt[32][72];       // [key][dh]
  __shared__ _Float16 Vt[64][56];       // transposed: [dh][key], 112B stride
  __shared__ _Float16 Ps[8][16][56];    // per-wave P (16 rows x 32 keys)
  const int b = blockIdx.z, h = blockIdx.y;
  const int tid  = threadIdx.x;
  const int wave = tid >> 5, lane = tid & 31;
  const int lo16 = lane & 15, hi8 = (lane & 16) ? 8 : 0;
  const int qTile = blockIdx.x * 128;
  const int q0 = qTile + wave * 16;

  const float* qp = Q + (size_t)(b * Lq + q0 + lo16) * 512 + h * 64;
  v16h qf0 = fragA_global(qp, hi8);
  v16h qf1 = fragA_global(qp + 32, hi8);

  float mrow[8], lrow[8];
  v8f o0 = {}, o1 = {}, o2 = {}, o3 = {};
#pragma unroll
  for (int r = 0; r < 8; ++r) { mrow[r] = -3.0e38f; lrow[r] = 0.f; }

  const int krow = tid >> 3, kcol = (tid & 7) * 8;   // K-tile staging map
  const int vdh = tid & 63, vkg = (tid >> 6) * 8;    // V-tile transpose map

  const int kend = causal ? ((qTile + 128 < Lk) ? qTile + 128 : Lk) : Lk;
  for (int kb = 0; kb < kend; kb += 32) {
    // stage K tile (vector) and V tile transposed (lane-coalesced gather)
    st8h(&Kt[krow][kcol],
         Kb + (size_t)(b * Lk + kb + krow) * 512 + h * 64 + kcol);
    {
      float tv[8];
#pragma unroll
      for (int j = 0; j < 8; ++j)
        tv[j] = Vb[(size_t)(b * Lk + kb + vkg + j) * 512 + h * 64 + vdh];
      pack8(&Vt[vdh][vkg], tv);
    }
    __syncthreads();

    // S = Q @ K^T : two 16-key subtiles x two 32-wide dh chunks
    v8f s0 = {}, s1 = {};
#pragma unroll
    for (int c = 0; c < 2; ++c) {
      v16h b0 = ld_frag(&Kt[lo16][c * 32], hi8);
      v16h b1 = ld_frag(&Kt[16 + lo16][c * 32], hi8);
      s0 = wmma16(c ? qf1 : qf0, b0, s0);
      s1 = wmma16(c ? qf1 : qf0, b1, s1);
    }

    // online softmax (rows per-VGPR, columns across 16-lane groups)
#pragma unroll
    for (int r = 0; r < 8; ++r) {
      int qi = q0 + r + hi8;
      float v0 = s0[r] * scale;
      float v1 = s1[r] * scale;
      if (causal) {
        if (kb + lo16 > qi)      v0 = -1.0e9f;
        if (kb + 16 + lo16 > qi) v1 = -1.0e9f;
      }
      float rm = fmaxf(v0, v1);
      rm = fmaxf(rm, __shfl_xor(rm, 1));
      rm = fmaxf(rm, __shfl_xor(rm, 2));
      rm = fmaxf(rm, __shfl_xor(rm, 4));
      rm = fmaxf(rm, __shfl_xor(rm, 8));
      float mnew  = fmaxf(mrow[r], rm);
      float alpha = __expf(mrow[r] - mnew);
      mrow[r] = mnew;
      float p0 = __expf(v0 - mnew);
      float p1 = __expf(v1 - mnew);
      float rs = p0 + p1;
      rs += __shfl_xor(rs, 1);
      rs += __shfl_xor(rs, 2);
      rs += __shfl_xor(rs, 4);
      rs += __shfl_xor(rs, 8);
      lrow[r] = lrow[r] * alpha + rs;
      o0[r] *= alpha; o1[r] *= alpha; o2[r] *= alpha; o3[r] *= alpha;
      Ps[wave][r + hi8][lo16]      = (_Float16)p0;
      Ps[wave][r + hi8][16 + lo16] = (_Float16)p1;
    }
    asm volatile("s_wait_dscnt 0" ::: "memory");  // P scratch RAW fence

    v16h pf = ld_frag(&Ps[wave][lo16][0], hi8);

    // O += P(16x32) @ V(32x64)
    {
      v16h bv0 = ld_frag(&Vt[lo16][0], hi8);
      v16h bv1 = ld_frag(&Vt[16 + lo16][0], hi8);
      v16h bv2 = ld_frag(&Vt[32 + lo16][0], hi8);
      v16h bv3 = ld_frag(&Vt[48 + lo16][0], hi8);
      o0 = wmma16(pf, bv0, o0);
      o1 = wmma16(pf, bv1, o1);
      o2 = wmma16(pf, bv2, o2);
      o3 = wmma16(pf, bv3, o3);
    }
    __syncthreads();
  }

#pragma unroll
  for (int r = 0; r < 8; ++r) {
    int qi = q0 + r + hi8;
    float inv = 1.f / lrow[r];
    float* op = O + (size_t)(b * Lq + qi) * 512 + h * 64;
    op[lo16]      = o0[r] * inv;
    op[16 + lo16] = o1[r] * inv;
    op[32 + lo16] = o2[r] * inv;
    op[48 + lo16] = o3[r] * inv;
  }
}

// ---------------------------------------------------------------------------
// LayerNorm over last dim (512); out = LN(x [+ res]) * g + b. grid = rows.
// ---------------------------------------------------------------------------
__global__ void layernorm_kernel(const float* __restrict__ x,
                                 const float* __restrict__ res,
                                 const float* __restrict__ g,
                                 const float* __restrict__ bb,
                                 float* __restrict__ out) {
  __shared__ float red[8];
  __shared__ float tot[2];
  const int row = blockIdx.x, tid = threadIdx.x;
  const float* xr = x + (size_t)row * 512;
  float v0 = xr[tid], v1 = xr[tid + 256];
  if (res) {
    const float* rr = res + (size_t)row * 512;
    v0 += rr[tid]; v1 += rr[tid + 256];
  }
  float s = v0 + v1;
  s += __shfl_xor(s, 1); s += __shfl_xor(s, 2); s += __shfl_xor(s, 4);
  s += __shfl_xor(s, 8); s += __shfl_xor(s, 16);
  if ((tid & 31) == 0) red[tid >> 5] = s;
  __syncthreads();
  if (tid == 0) {
    float t = 0.f;
    for (int i = 0; i < 8; ++i) t += red[i];
    tot[0] = t;
  }
  __syncthreads();
  float mu = tot[0] * (1.f / 512.f);
  float d0 = v0 - mu, d1 = v1 - mu;
  float ss = d0 * d0 + d1 * d1;
  ss += __shfl_xor(ss, 1); ss += __shfl_xor(ss, 2); ss += __shfl_xor(ss, 4);
  ss += __shfl_xor(ss, 8); ss += __shfl_xor(ss, 16);
  if ((tid & 31) == 0) red[tid >> 5] = ss;
  __syncthreads();
  if (tid == 0) {
    float t = 0.f;
    for (int i = 0; i < 8; ++i) t += red[i];
    tot[1] = t;
  }
  __syncthreads();
  float rstd = rsqrtf(tot[1] * (1.f / 512.f) + 1e-5f);
  out[(size_t)row * 512 + tid]       = d0 * rstd * g[tid] + bb[tid];
  out[(size_t)row * 512 + tid + 256] = d1 * rstd * g[tid + 256] + bb[tid + 256];
}

// ---------------------------------------------------------------------------
// BatchNorm stats over (B,L) per channel. grid = 512 channels.
// ---------------------------------------------------------------------------
__global__ void bn_stats_kernel(const float* __restrict__ y,
                                float* __restrict__ mv, int M) {
  __shared__ float r1[8], r2[8];
  const int c = blockIdx.x, tid = threadIdx.x;
  float s = 0.f, s2 = 0.f;
  for (int i = tid; i < M; i += 256) {
    float v = y[(size_t)i * 512 + c];
    s += v; s2 += v * v;
  }
  s  += __shfl_xor(s, 1);  s  += __shfl_xor(s, 2);  s  += __shfl_xor(s, 4);
  s  += __shfl_xor(s, 8);  s  += __shfl_xor(s, 16);
  s2 += __shfl_xor(s2, 1); s2 += __shfl_xor(s2, 2); s2 += __shfl_xor(s2, 4);
  s2 += __shfl_xor(s2, 8); s2 += __shfl_xor(s2, 16);
  if ((tid & 31) == 0) { r1[tid >> 5] = s; r2[tid >> 5] = s2; }
  __syncthreads();
  if (tid == 0) {
    float t1 = 0.f, t2 = 0.f;
    for (int i = 0; i < 8; ++i) { t1 += r1[i]; t2 += r2[i]; }
    float mu = t1 / (float)M;
    mv[c]       = mu;
    mv[512 + c] = t2 / (float)M - mu * mu;
  }
}

// ---------------------------------------------------------------------------
// BN apply + ELU + maxpool(window 3, stride 2, pad -inf). out length L/2.
// ---------------------------------------------------------------------------
__global__ void bn_elu_pool_kernel(const float* __restrict__ y,
                                   const float* __restrict__ mv,
                                   const float* __restrict__ g,
                                   const float* __restrict__ bb,
                                   float* __restrict__ out, int L) {
  const int idx = blockIdx.x * 256 + threadIdx.x;
  const int c = idx & 511;
  const int rest = idx >> 9;
  const int Lh = L >> 1;
  const int lo = rest % Lh;
  const int b  = rest / Lh;
  const float mu = mv[c];
  const float rstd = rsqrtf(mv[512 + c] + 1e-5f);
  const float gc = g[c], bc = bb[c];
  float best = -3.0e38f;
#pragma unroll
  for (int t = -1; t <= 1; ++t) {
    int l = 2 * lo + t;
    if (l < 0 || l >= L) continue;
    float v = (y[(size_t)(b * L + l) * 512 + c] - mu) * rstd * gc + bc;
    v = v > 0.f ? v : (__expf(v) - 1.f);
    best = fmaxf(best, v);
  }
  out[(size_t)(b * Lh + lo) * 512 + c] = best;
}

// ---------------------------------------------------------------------------
// Embedding: circular token conv (3,CIN,512) + sinusoidal PE + mark @ W.
// ---------------------------------------------------------------------------
__global__ void embed_kernel(const float* __restrict__ x,
                             const float* __restrict__ xm,
                             const float* __restrict__ cw,
                             const float* __restrict__ mw,
                             float* __restrict__ out, int L, int CIN) {
  const int idx = blockIdx.x * 256 + threadIdx.x;  // (b*L+l)*512+c
  const int c = idx & 511;
  const int rest = idx >> 9;
  const int l = rest % L;
  const int b = rest / L;
  float acc = 0.f;
#pragma unroll
  for (int t = 0; t < 3; ++t) {
    int l2 = l + t - 1;
    if (l2 < 0) l2 += L;
    if (l2 >= L) l2 -= L;
    const float* xp = x + (size_t)(b * L + l2) * CIN;
    for (int ci = 0; ci < CIN; ++ci) acc += xp[ci] * cw[(t * CIN + ci) * 512 + c];
  }
  const float* mp = xm + (size_t)(b * L + l) * 4;
#pragma unroll
  for (int j = 0; j < 4; ++j) acc += mp[j] * mw[j * 512 + c];
  const int i2 = c & ~1;
  float freq = __expf(-(float)i2 * (9.210340371976184f / 512.f));  // ln(1e4)
  float ang = (float)l * freq;
  acc += (c & 1) ? __cosf(ang) : __sinf(ang);
  out[idx] = acc;
}

// ---------------------------------------------------------------------------
// Final projection: out[b, lo, c] = dec[b, 512+lo, :] @ proj_w + proj_b (N=7)
// ---------------------------------------------------------------------------
__global__ void proj_kernel(const float* __restrict__ x,
                            const float* __restrict__ w,
                            const float* __restrict__ bb,
                            float* __restrict__ out) {
  const int idx = blockIdx.x * 256 + threadIdx.x;
  if (idx >= 8 * 512 * 7) return;
  const int c = idx % 7;
  const int rest = idx / 7;
  const int lo = rest & 511;
  const int b = rest >> 9;
  const float4* xr = (const float4*)(x + (size_t)(b * 1024 + 512 + lo) * 512);
  float acc = bb[c];
  for (int kk = 0; kk < 128; ++kk) {
    float4 xv = xr[kk];
    const float* wp = w + (size_t)(4 * kk) * 7 + c;
    acc += xv.x * wp[0] + xv.y * wp[7] + xv.z * wp[14] + xv.w * wp[21];
  }
  out[idx] = acc;
}

// ---------------------------------------------------------------------------
// Orchestration. Param leaf order = jax pytree flatten (sorted dict keys):
// 0 x_enc, 1 x_mark_enc, 2 x_dec, 3 x_mark_dec, then:
// conv_layers[0]{b=4,bn_b=5,bn_g=6,w=7}, dec_emb{conv_w=8,mark_w=9},
// dec_layers[0]{b1=10,b2=11, ca{bk=12,bo=13,bq=14,bv=15,wk=16,wo=17,wq=18,wv=19},
//   ln1_b=20,ln1_g=21,ln2_b=22,ln2_g=23,ln3_b=24,ln3_g=25,
//   sa{bk=26,bo=27,bq=28,bv=29,wk=30,wo=31,wq=32,wv=33}, w1=34,w2=35},
// dec_norm_b=36, dec_norm_g=37, enc_emb{conv_w=38,mark_w=39},
// enc_layers[i]{attn{bk,bo,bq,bv,wk,wo,wq,wv}=base..+7, b1=+8,b2=+9,
//   ln1_b=+10,ln1_g=+11,ln2_b=+12,ln2_g=+13, w1=+14,w2=+15} base=40,56,
// enc_norm_b=72, enc_norm_g=73, proj_b=74, proj_w=75.
// ---------------------------------------------------------------------------
extern "C" void kernel_launch(void* const* d_in, const int* in_sizes, int n_in,
                              void* d_out, int out_size, void* d_ws, size_t ws_size,
                              hipStream_t stream) {
  auto P = [&](int i) { return (const float*)d_in[i]; };
  const int B = 8, L0 = 1024, D = 512, H = 8, FF = 2048;
  const size_t SZ = (size_t)B * L0 * D;

  float* enc = (float*)d_ws;
  float* dec = enc + SZ;
  float* q   = dec + SZ;
  float* k   = q + SZ;
  float* v   = k + SZ;
  float* t1  = v + SZ;
  float* t2  = t1 + SZ;
  float* ff  = t2 + SZ;                 // B*L0*FF
  float* cy  = ff + (size_t)B * L0 * FF;
  float* mv  = cy + SZ;                 // 1024 floats of BN stats

  const dim3 blk(256);

  // ---- encoder ----
  embed_kernel<<<(B * L0 * D) / 256, blk, 0, stream>>>(P(0), P(1), P(38), P(39),
                                                       enc, L0, 7);
  int Lcur = L0;
  const int encBase[2] = {40, 56};
  for (int i = 0; i < 2; ++i) {
    const int eb = encBase[i];
    const int M = B * Lcur;
    const dim3 gg(M / 128, D / 64);
    gemm_kernel<<<gg, blk, 0, stream>>>(enc, P(eb + 6), P(eb + 2), q, M, D, D, 0);
    gemm_kernel<<<gg, blk, 0, stream>>>(enc, P(eb + 4), P(eb + 0), k, M, D, D, 0);
    gemm_kernel<<<gg, blk, 0, stream>>>(enc, P(eb + 7), P(eb + 3), v, M, D, D, 0);
    attn_kernel<<<dim3(Lcur / 128, H, B), blk, 0, stream>>>(q, k, v, t1, Lcur,
                                                            Lcur, 0.125f, 0);
    gemm_kernel<<<gg, blk, 0, stream>>>(t1, P(eb + 5), P(eb + 1), t2, M, D, D, 0);
    layernorm_kernel<<<M, blk, 0, stream>>>(enc, t2, P(eb + 11), P(eb + 10), enc);
    gemm_kernel<<<dim3(M / 128, FF / 64), blk, 0, stream>>>(
        enc, P(eb + 14), P(eb + 8), ff, M, FF, D, 1);
    gemm_kernel<<<dim3(M / 128, D / 64), blk, 0, stream>>>(
        ff, P(eb + 15), P(eb + 9), t2, M, D, FF, 0);
    layernorm_kernel<<<M, blk, 0, stream>>>(enc, t2, P(eb + 13), P(eb + 12), enc);
    if (i == 0) {  // distilling conv + BN + ELU + maxpool
      conv_gemm_kernel<<<dim3(Lcur / 128, D / 64, B), blk, 0, stream>>>(
          enc, P(7), P(4), cy, Lcur);
      bn_stats_kernel<<<D, blk, 0, stream>>>(cy, mv, B * Lcur);
      bn_elu_pool_kernel<<<(B * (Lcur / 2) * D) / 256, blk, 0, stream>>>(
          cy, mv, P(6), P(5), enc, Lcur);
      Lcur /= 2;
    }
  }
  layernorm_kernel<<<B * Lcur, blk, 0, stream>>>(enc, nullptr, P(73), P(72), enc);
  const int Lenc = Lcur;  // 512

  // ---- decoder ----
  embed_kernel<<<(B * L0 * D) / 256, blk, 0, stream>>>(P(2), P(3), P(8), P(9),
                                                       dec, L0, 7);
  const int M = B * L0;
  const dim3 gd(M / 128, D / 64);
  // self-attention (causal)
  gemm_kernel<<<gd, blk, 0, stream>>>(dec, P(32), P(28), q, M, D, D, 0);
  gemm_kernel<<<gd, blk, 0, stream>>>(dec, P(30), P(26), k, M, D, D, 0);
  gemm_kernel<<<gd, blk, 0, stream>>>(dec, P(33), P(29), v, M, D, D, 0);
  attn_kernel<<<dim3(L0 / 128, H, B), blk, 0, stream>>>(q, k, v, t1, L0, L0,
                                                        0.125f, 1);
  gemm_kernel<<<gd, blk, 0, stream>>>(t1, P(31), P(27), t2, M, D, D, 0);
  layernorm_kernel<<<M, blk, 0, stream>>>(dec, t2, P(21), P(20), dec);
  // cross-attention (softmax(FACTOR * scores))
  const int Me = B * Lenc;
  gemm_kernel<<<gd, blk, 0, stream>>>(dec, P(18), P(14), q, M, D, D, 0);
  gemm_kernel<<<dim3(Me / 128, D / 64), blk, 0, stream>>>(enc, P(16), P(12), k,
                                                          Me, D, D, 0);
  gemm_kernel<<<dim3(Me / 128, D / 64), blk, 0, stream>>>(enc, P(19), P(15), v,
                                                          Me, D, D, 0);
  attn_kernel<<<dim3(L0 / 128, H, B), blk, 0, stream>>>(q, k, v, t1, L0, Lenc,
                                                        5.0f, 0);
  gemm_kernel<<<gd, blk, 0, stream>>>(t1, P(17), P(13), t2, M, D, D, 0);
  layernorm_kernel<<<M, blk, 0, stream>>>(dec, t2, P(23), P(22), dec);
  // FFN
  gemm_kernel<<<dim3(M / 128, FF / 64), blk, 0, stream>>>(dec, P(34), P(10), ff,
                                                          M, FF, D, 1);
  gemm_kernel<<<gd, blk, 0, stream>>>(ff, P(35), P(11), t2, M, D, FF, 0);
  layernorm_kernel<<<M, blk, 0, stream>>>(dec, t2, P(25), P(24), dec);
  layernorm_kernel<<<M, blk, 0, stream>>>(dec, nullptr, P(37), P(36), dec);
  // final projection, last PRED_LEN=512 positions
  proj_kernel<<<(8 * 512 * 7 + 255) / 256, blk, 0, stream>>>(dec, P(75), P(74),
                                                             (float*)d_out);
  (void)in_sizes; (void)n_in; (void)out_size; (void)ws_size;
}
